// RDSSM_module_77627238908759
// MI455X (gfx1250) — matrix operand order repaired and checked by reading, not compiled, not run on hardware
//
#include <hip/hip_runtime.h>
#include <hip/hip_bf16.h>
#include <math.h>

// ---------------------------------------------------------------------------
// RDSSM (VRNN) forward: kld_loss, nll_loss.
// Phase 0: convert all weights f32->f16 into ws (L2-resident, ~2.2MB).
// Phase 1: 64 WGs x 16 rows, full T=256 scan in-LDS, all GEMMs via
//          v_wmma_f32_16x16x32_f16 with paired output tiles (A-frag reuse).
// Phase 2: decoder over 262144 rows, 32 rows/WG (weight amortization x2).
// Phase 3: finalize two scalars.
// ---------------------------------------------------------------------------

typedef __attribute__((ext_vector_type(16))) _Float16 v16h;
typedef __attribute__((ext_vector_type(8)))  _Float16 v8h;
typedef __attribute__((ext_vector_type(8)))  float    v8f;

#define B_    128
#define T_    256
#define XD_   32
#define H_    256
#define Z_    64
#define IW_   8
#define M1_   (B_*IW_)      /* 1024 rows in the scan */
#define MT_   16            /* rows per workgroup, phase 1 */
#define MT2_  32            /* rows per workgroup, phase 2 */
#define LOG2PI_ 1.8378770664093453f

// f16 weight pool offsets (elements)
#define WO_DEC0      0
#define WO_DEC2      131072
#define WO_DECMEAN   196608
#define WO_DECSTD    204800
#define WO_ENC0      212992
#define WO_ENC2      409600
#define WO_ENCMEAN   475136
#define WO_ENCSTD    491520
#define WO_GRUWHH    507904
#define WO_GRUWIH    704512
#define WO_PHIX      901120
#define WO_PHIZ      909312
#define WO_PRIOR0    925696
#define WO_PRIORMEAN 1056768
#define WO_PRIORSTD  1073152
#define W16_TOTAL    1089536

#define OFF_ACC 0
#define OFF_W16 256
#define OFF_HS  (OFF_W16 + (size_t)W16_TOTAL*2)
#define HS_BYTES ((size_t)T_*M1_*H_*2)
#define OFF_ZHS (OFF_HS + HS_BYTES)
// total ws required = OFF_ZHS + HS_BYTES  (~271 MB)

// ---- math helpers ---------------------------------------------------------
__device__ __forceinline__ float sigmoidf_(float x) { return 1.f / (1.f + expf(-x)); }
__device__ __forceinline__ float softplusf_(float x) { return x > 20.f ? x : log1pf(expf(x)); }

// ---- WMMA fragment loads (contiguous k order, 16-bit, wave32) -------------
// A (16xK tile slice): lane holds row lane&15; elems j <- K=kb+16*half+j (contiguous)
__device__ __forceinline__ v16h frag_a(const _Float16* rowptr, int kb, int half) {
  return *(const v16h*)(rowptr + kb + 16 * half);
}
// B (Kx16 tile slice): lane holds col lane&15; elems j <- K=kb+16*half+j (contiguous)
__device__ __forceinline__ v16h frag_b(const _Float16* rowptr, int kb, int half) {
  return *(const v16h*)(rowptr + kb + 16 * half);
}
__device__ __forceinline__ v8f wmma16(v16h a, v16h b, v8f c) {
  return __builtin_amdgcn_wmma_f32_16x16x32_f16(false, a, false, b, (short)0, c, false, false);
}

// Two 16x16 output tiles sharing one A-fragment stream:
//   Ca += A * W[n0a:n0a+16, kOff:kOff+K]^T ; Cb += A * W[n0b:n0b+16, ...]^T
__device__ __forceinline__ void gemm_acc2(v8f& ca, v8f& cb, const _Float16* aLds, int lda,
                                          const _Float16* w, int ldw, int kOff,
                                          int n0a, int n0b, int K, int lane) {
  const int r = lane & 15, half = lane >> 4;
  const _Float16* arow = aLds + (size_t)r * lda;
  const _Float16* wra = w + (size_t)(n0a + r) * ldw + kOff;
  const _Float16* wrb = w + (size_t)(n0b + r) * ldw + kOff;
  #pragma unroll 4
  for (int kb = 0; kb < K; kb += 32) {
    v16h af = frag_a(arow, kb, half);
    ca = wmma16(af, frag_b(wra, kb, half), ca);
    cb = wmma16(af, frag_b(wrb, kb, half), cb);
  }
}

// D-tile epilogue: element i of lane -> (row i+8*half, col lane&15)
__device__ __forceinline__ void store_tile(_Float16* out, int ldo, int n0, v8f c,
                                           const float* bias, bool do_relu, int lane) {
  const int col = lane & 15, half = lane >> 4;
  const float bb = bias[n0 + col];
  #pragma unroll
  for (int i = 0; i < 8; i++) {
    float v = c[i] + bb;
    if (do_relu) v = fmaxf(v, 0.f);
    out[(size_t)(i + 8 * half) * ldo + n0 + col] = (_Float16)v;
  }
}
__device__ __forceinline__ void store_tile_b2(_Float16* out, int ldo, int n0, v8f c,
                                              const float* b1, const float* b2, int lane) {
  const int col = lane & 15, half = lane >> 4;
  const float bb = b1[n0 + col] + b2[n0 + col];
  #pragma unroll
  for (int i = 0; i < 8; i++)
    out[(size_t)(i + 8 * half) * ldo + n0 + col] = (_Float16)(c[i] + bb);
}
// relu+bias, write to LDS and global (zh / zhs[t])
__device__ __forceinline__ void store_tile_dual(_Float16* lds, _Float16* glob, int ldo, int n0,
                                                v8f c, const float* bias, int lane) {
  const int col = lane & 15, half = lane >> 4;
  const float bb = bias[n0 + col];
  #pragma unroll
  for (int i = 0; i < 8; i++) {
    _Float16 hv = (_Float16)fmaxf(c[i] + bb, 0.f);
    size_t o = (size_t)(i + 8 * half) * ldo + n0 + col;
    lds[o] = hv;
    glob[o] = hv;
  }
}

// ---- small kernels --------------------------------------------------------
__global__ void k_init(float* acc) { if (threadIdx.x < 2) acc[threadIdx.x] = 0.f; }

__global__ void k_cvt(const float* __restrict__ src, _Float16* __restrict__ dst, int n) {
  int i = blockIdx.x * blockDim.x + threadIdx.x;
  if (i < n) dst[i] = (_Float16)src[i];
}

__global__ void k_fin(const float* __restrict__ acc, float* __restrict__ out) {
  out[0] = acc[0] / (float)(B_ * IW_);
  out[1] = -acc[1] / (float)(B_ * IW_);
}

// ---- Phase 1: recurrent scan ---------------------------------------------
__global__ __launch_bounds__(256)
void phase1_scan(const float* __restrict__ X, const float* __restrict__ eps,
                 const _Float16* __restrict__ w16,
                 const float* __restrict__ b_phix, const float* __restrict__ b_phiz,
                 const float* __restrict__ b_prior0,
                 const float* __restrict__ b_pmean, const float* __restrict__ b_pstd,
                 const float* __restrict__ b_enc0, const float* __restrict__ b_enc2,
                 const float* __restrict__ b_qmean, const float* __restrict__ b_qstd,
                 const float* __restrict__ b_ih, const float* __restrict__ b_hh,
                 _Float16* __restrict__ hs, _Float16* __restrict__ zhs,
                 float* __restrict__ acc) {
  __shared__ __align__(32) _Float16 sh_h  [MT_ * H_];
  __shared__ __align__(32) _Float16 sh_xh [MT_ * H_];
  __shared__ __align__(32) _Float16 sh_zh [MT_ * H_];
  __shared__ __align__(32) _Float16 sh_t0 [MT_ * H_];   // ph, eh, gn_i
  __shared__ __align__(32) _Float16 sh_t1 [MT_ * H_];   // eh2, gn_h
  __shared__ __align__(32) _Float16 sh_grz[MT_ * 512];
  __shared__ __align__(32) _Float16 sh_zn [MT_ * Z_];
  __shared__ __align__(32) _Float16 sh_xin[MT_ * XD_];
  __shared__ float sh_red[256];

  const int tid = threadIdx.x, lane = tid & 31, wave = tid >> 5;
  const int r0 = blockIdx.x * MT_;
  const int na = wave * 16, nb = (wave + 8) * 16;   // paired col tiles (N=256)

  const _Float16* w_phix  = w16 + WO_PHIX;
  const _Float16* w_phiz  = w16 + WO_PHIZ;
  const _Float16* w_pri0  = w16 + WO_PRIOR0;
  const _Float16* w_pmean = w16 + WO_PRIORMEAN;
  const _Float16* w_pstd  = w16 + WO_PRIORSTD;
  const _Float16* w_enc0  = w16 + WO_ENC0;
  const _Float16* w_enc2  = w16 + WO_ENC2;
  const _Float16* w_qmean = w16 + WO_ENCMEAN;
  const _Float16* w_qstd  = w16 + WO_ENCSTD;
  const _Float16* w_ih    = w16 + WO_GRUWIH;
  const _Float16* w_hh    = w16 + WO_GRUWHH;

  float klacc = 0.f;

  // ---------------- prologue: xh0, zh = relu(b_phiz), h0 = GRU(xh0, 0) -----
  for (int e = tid; e < MT_ * XD_; e += 256) {
    int lr = e >> 5, c = e & 31;
    int b = (r0 + lr) >> 3;
    sh_xin[e] = (_Float16)X[((size_t)b * T_ + 0) * XD_ + c];
  }
  for (int e = tid; e < MT_ * H_; e += 256)
    sh_zh[e] = (_Float16)fmaxf(b_phiz[e & (H_ - 1)], 0.f);
  __syncthreads();
  {
    v8f c0{}, c1{};
    gemm_acc2(c0, c1, sh_xin, XD_, w_phix, XD_, 0, na, nb, XD_, lane);
    store_tile(sh_xh, H_, na, c0, b_phix, true, lane);
    store_tile(sh_xh, H_, nb, c1, b_phix, true, lane);
  }
  __syncthreads();
  #pragma unroll
  for (int p = 0; p < 2; ++p) {                      // r,z gates (h=0 -> gh=b_hh)
    int ma = (wave + p * 16) * 16, mb = (wave + p * 16 + 8) * 16;
    v8f c0{}, c1{};
    gemm_acc2(c0, c1, sh_xh, H_, w_ih, H_, 0, ma, mb, H_, lane);
    store_tile_b2(sh_grz, 512, ma, c0, b_ih, b_hh, lane);
    store_tile_b2(sh_grz, 512, mb, c1, b_ih, b_hh, lane);
  }
  {
    v8f c0{}, c1{};                                  // gi_n
    gemm_acc2(c0, c1, sh_xh, H_, w_ih + 512 * H_, H_, 0, na, nb, H_, lane);
    store_tile(sh_t0, H_, na, c0, b_ih + 512, false, lane);
    store_tile(sh_t0, H_, nb, c1, b_ih + 512, false, lane);
  }
  __syncthreads();
  for (int e = tid; e < MT_ * H_; e += 256) {
    int lr = e >> 8, c = e & 255;
    float r  = sigmoidf_((float)sh_grz[lr * 512 + c]);
    float zg = sigmoidf_((float)sh_grz[lr * 512 + 256 + c]);
    float n  = tanhf((float)sh_t0[e] + r * b_hh[512 + c]);
    sh_h[e] = (_Float16)((1.f - zg) * n);
  }
  __syncthreads();

  // ---------------- scan ----------------------------------------------------
  for (int t = 0; t < T_; ++t) {
    v8f pmu{}, pstd{};
    // hs[t] = h (pre-update), load X[:,t]
    {
      v8h* dst = (v8h*)(hs + ((size_t)t * M1_ + r0) * H_);
      const v8h* src = (const v8h*)sh_h;
      for (int j = tid; j < MT_ * H_ / 8; j += 256) dst[j] = src[j];
    }
    for (int e = tid; e < MT_ * XD_; e += 256) {
      int lr = e >> 5, c = e & 31;
      int b = (r0 + lr) >> 3;
      sh_xin[e] = (_Float16)X[((size_t)b * T_ + t) * XD_ + c];
    }
    __syncthreads();
    // xh = relu(phi_x(x))
    {
      v8f c0{}, c1{};
      gemm_acc2(c0, c1, sh_xin, XD_, w_phix, XD_, 0, na, nb, XD_, lane);
      store_tile(sh_xh, H_, na, c0, b_phix, true, lane);
      store_tile(sh_xh, H_, nb, c1, b_phix, true, lane);
    }
    __syncthreads();
    // ph = relu(prior0 @ [zh, h]) -> t0
    {
      v8f c0{}, c1{};
      gemm_acc2(c0, c1, sh_zh, H_, w_pri0, 512, 0,   na, nb, H_, lane);
      gemm_acc2(c0, c1, sh_h,  H_, w_pri0, 512, 256, na, nb, H_, lane);
      store_tile(sh_t0, H_, na, c0, b_prior0, true, lane);
      store_tile(sh_t0, H_, nb, c1, b_prior0, true, lane);
    }
    __syncthreads();
    // prior heads: waves 0-3 keep p_mu/p_std in registers (cols 16w..16w+15)
    if (wave < 4) {
      const int n0 = wave * 16, r = lane & 15, half = lane >> 4;
      const _Float16* arow = sh_t0 + (size_t)r * H_;
      const _Float16* wm = w_pmean + (size_t)(n0 + r) * H_;
      const _Float16* wsd = w_pstd + (size_t)(n0 + r) * H_;
      for (int kb = 0; kb < H_; kb += 32) {
        v16h af = frag_a(arow, kb, half);
        pmu  = wmma16(af, frag_b(wm,  kb, half), pmu);
        pstd = wmma16(af, frag_b(wsd, kb, half), pstd);
      }
      const int col = n0 + (lane & 15);
      #pragma unroll
      for (int i = 0; i < 8; i++) {
        pmu[i]  = pmu[i] + b_pmean[col];
        pstd[i] = softplusf_(pstd[i] + b_pstd[col]);
      }
    }
    __syncthreads();
    // eh = relu(enc0 @ [xh, zh, h]) -> t0
    {
      v8f c0{}, c1{};
      gemm_acc2(c0, c1, sh_xh, H_, w_enc0, 768, 0,   na, nb, H_, lane);
      gemm_acc2(c0, c1, sh_zh, H_, w_enc0, 768, 256, na, nb, H_, lane);
      gemm_acc2(c0, c1, sh_h,  H_, w_enc0, 768, 512, na, nb, H_, lane);
      store_tile(sh_t0, H_, na, c0, b_enc0, true, lane);
      store_tile(sh_t0, H_, nb, c1, b_enc0, true, lane);
    }
    __syncthreads();
    // eh2 = relu(enc2 @ eh) -> t1
    {
      v8f c0{}, c1{};
      gemm_acc2(c0, c1, sh_t0, H_, w_enc2, H_, 0, na, nb, H_, lane);
      store_tile(sh_t1, H_, na, c0, b_enc2, true, lane);
      store_tile(sh_t1, H_, nb, c1, b_enc2, true, lane);
    }
    __syncthreads();
    // enc heads + z_new + KL (waves 0-3; same col mapping as prior heads)
    if (wave < 4) {
      const int n0 = wave * 16, r = lane & 15, half = lane >> 4;
      const _Float16* arow = sh_t1 + (size_t)r * H_;
      const _Float16* wm = w_qmean + (size_t)(n0 + r) * H_;
      const _Float16* wsd = w_qstd + (size_t)(n0 + r) * H_;
      v8f qmu{}, qstd{};
      for (int kb = 0; kb < H_; kb += 32) {
        v16h af = frag_a(arow, kb, half);
        qmu  = wmma16(af, frag_b(wm,  kb, half), qmu);
        qstd = wmma16(af, frag_b(wsd, kb, half), qstd);
      }
      const int col = n0 + (lane & 15);
      #pragma unroll
      for (int i = 0; i < 8; i++) {
        int lr = i + 8 * half;
        int grow = r0 + lr;
        float qm = qmu[i] + b_qmean[col];
        float qs = softplusf_(qstd[i] + b_qstd[col]);
        float ev = eps[((size_t)t * M1_ + grow) * Z_ + col];
        float zn = qm + qs * ev;
        sh_zn[lr * Z_ + col] = (_Float16)zn;
        float dm = qm - pmu[i];
        klacc += logf(pstd[i]) - logf(qs) +
                 (qs * qs + dm * dm) / (2.f * pstd[i] * pstd[i]) - 0.5f;
      }
    }
    __syncthreads();
    // zh_next = relu(phi_z(z_new)) -> sh_zh and zhs[t]
    {
      v8f c0{}, c1{};
      gemm_acc2(c0, c1, sh_zn, Z_, w_phiz, Z_, 0, na, nb, Z_, lane);
      _Float16* g = zhs + ((size_t)t * M1_ + r0) * H_;
      store_tile_dual(sh_zh, g, H_, na, c0, b_phiz, lane);
      store_tile_dual(sh_zh, g, H_, nb, c1, b_phiz, lane);
    }
    // GRU gates: grz = xh@Wih[rz] + h@Whh[rz] + b; gn_i -> t0; gn_h -> t1
    #pragma unroll
    for (int p = 0; p < 2; ++p) {
      int ma = (wave + p * 16) * 16, mb = (wave + p * 16 + 8) * 16;
      v8f c0{}, c1{};
      gemm_acc2(c0, c1, sh_xh, H_, w_ih, H_, 0, ma, mb, H_, lane);
      gemm_acc2(c0, c1, sh_h,  H_, w_hh, H_, 0, ma, mb, H_, lane);
      store_tile_b2(sh_grz, 512, ma, c0, b_ih, b_hh, lane);
      store_tile_b2(sh_grz, 512, mb, c1, b_ih, b_hh, lane);
    }
    {
      v8f c0{}, c1{};
      gemm_acc2(c0, c1, sh_xh, H_, w_ih + 512 * H_, H_, 0, na, nb, H_, lane);
      store_tile(sh_t0, H_, na, c0, b_ih + 512, false, lane);
      store_tile(sh_t0, H_, nb, c1, b_ih + 512, false, lane);
    }
    {
      v8f c0{}, c1{};
      gemm_acc2(c0, c1, sh_h, H_, w_hh + 512 * H_, H_, 0, na, nb, H_, lane);
      store_tile(sh_t1, H_, na, c0, b_hh + 512, false, lane);
      store_tile(sh_t1, H_, nb, c1, b_hh + 512, false, lane);
    }
    __syncthreads();
    // h update
    for (int e = tid; e < MT_ * H_; e += 256) {
      int lr = e >> 8, c = e & 255;
      float r  = sigmoidf_((float)sh_grz[lr * 512 + c]);
      float zg = sigmoidf_((float)sh_grz[lr * 512 + 256 + c]);
      float n  = tanhf((float)sh_t0[e] + r * (float)sh_t1[e]);
      float hv = (float)sh_h[e];
      sh_h[e] = (_Float16)((1.f - zg) * n + zg * hv);
    }
    __syncthreads();
  }

  sh_red[tid] = klacc;
  __syncthreads();
  if (tid == 0) {
    float s = 0.f;
    for (int i = 0; i < 256; i++) s += sh_red[i];
    atomicAdd(&acc[0], s);
  }
}

// ---- Phase 2: decoder (32 rows / WG) --------------------------------------
__global__ __launch_bounds__(256)
void phase2_dec(const float* __restrict__ X, const _Float16* __restrict__ w16,
                const float* __restrict__ b_dec0, const float* __restrict__ b_dec2,
                const float* __restrict__ b_dmean, const float* __restrict__ b_dstd,
                const _Float16* __restrict__ hs, const _Float16* __restrict__ zhs,
                float* __restrict__ acc) {
  __shared__ __align__(32) _Float16 sh_zh[MT2_ * H_];  // zh, then dh2
  __shared__ __align__(32) _Float16 sh_h [MT2_ * H_];
  __shared__ __align__(32) _Float16 sh_t0[MT2_ * H_];
  __shared__ float sh_red[256];

  const int tid = threadIdx.x, lane = tid & 31, wave = tid >> 5;
  const size_t s0 = (size_t)blockIdx.x * MT2_;
  const int na = wave * 16, nb = (wave + 8) * 16;

  const _Float16* w_dec0  = w16 + WO_DEC0;
  const _Float16* w_dec2  = w16 + WO_DEC2;
  const _Float16* w_dmean = w16 + WO_DECMEAN;
  const _Float16* w_dstd  = w16 + WO_DECSTD;

  {
    const v8h* hsrc = (const v8h*)(hs + s0 * H_);
    const v8h* zsrc = (const v8h*)(zhs + s0 * H_);
    v8h* hdst = (v8h*)sh_h; v8h* zdst = (v8h*)sh_zh;
    for (int j = tid; j < MT2_ * H_ / 8; j += 256) { hdst[j] = hsrc[j]; zdst[j] = zsrc[j]; }
  }
  __syncthreads();
  // dh = relu(dec0 @ [zh, h]) -> t0   (2 row tiles x paired col tiles)
  #pragma unroll
  for (int rt = 0; rt < 2; ++rt) {
    v8f c0{}, c1{};
    gemm_acc2(c0, c1, sh_zh + rt * 16 * H_, H_, w_dec0, 512, 0,   na, nb, H_, lane);
    gemm_acc2(c0, c1, sh_h  + rt * 16 * H_, H_, w_dec0, 512, 256, na, nb, H_, lane);
    store_tile(sh_t0 + rt * 16 * H_, H_, na, c0, b_dec0, true, lane);
    store_tile(sh_t0 + rt * 16 * H_, H_, nb, c1, b_dec0, true, lane);
  }
  __syncthreads();
  // dh2 = relu(dec2 @ dh) -> sh_zh (zh dead after dec0)
  #pragma unroll
  for (int rt = 0; rt < 2; ++rt) {
    v8f c0{}, c1{};
    gemm_acc2(c0, c1, sh_t0 + rt * 16 * H_, H_, w_dec2, H_, 0, na, nb, H_, lane);
    store_tile(sh_zh + rt * 16 * H_, H_, na, c0, b_dec2, true, lane);
    store_tile(sh_zh + rt * 16 * H_, H_, nb, c1, b_dec2, true, lane);
  }
  __syncthreads();
  float lp = 0.f;
  if (wave < 4) {   // 2 row tiles x 2 col tiles (X_DIM=32); mu+std share A
    const int ct = wave & 1, rt = wave >> 1;
    const int n0 = ct * 16, r = lane & 15, half = lane >> 4;
    const _Float16* arow = sh_zh + (size_t)(rt * 16 + r) * H_;
    const _Float16* wm = w_dmean + (size_t)(n0 + r) * H_;
    const _Float16* wsd = w_dstd + (size_t)(n0 + r) * H_;
    v8f cmu{}, cstd{};
    for (int kb = 0; kb < H_; kb += 32) {
      v16h af = frag_a(arow, kb, half);
      cmu  = wmma16(af, frag_b(wm,  kb, half), cmu);
      cstd = wmma16(af, frag_b(wsd, kb, half), cstd);
    }
    const int col = n0 + (lane & 15);
    #pragma unroll
    for (int i = 0; i < 8; i++) {
      size_t s = s0 + rt * 16 + i + 8 * half;    // s = t*1024 + row
      int t = (int)(s >> 10);
      int b = ((int)(s & 1023)) >> 3;
      float xv = X[((size_t)b * T_ + t) * XD_ + col];
      float xm = cmu[i] + b_dmean[col];
      float xs = softplusf_(cstd[i] + b_dstd[col]) + 1.0f;
      float d = xv - xm;
      lp += -(d * d) / (2.f * xs * xs) - logf(xs) - 0.5f * LOG2PI_;
    }
  }
  sh_red[tid] = lp;
  __syncthreads();
  if (tid == 0) {
    float s = 0.f;
    for (int i = 0; i < 256; i++) s += sh_red[i];
    atomicAdd(&acc[1], s);
  }
}

// ---------------------------------------------------------------------------
extern "C" void kernel_launch(void* const* d_in, const int* in_sizes, int n_in,
                              void* d_out, int out_size, void* d_ws, size_t ws_size,
                              hipStream_t stream) {
  (void)in_sizes; (void)n_in; (void)out_size; (void)ws_size;
  const float* X   = (const float*)d_in[0];
  const float* eps = (const float*)d_in[1];
  // params flattened in sorted-key order, (w, b) per entry:
  const float* dec0_w   = (const float*)d_in[2];
  const float* dec0_b   = (const float*)d_in[3];
  const float* dec2_w   = (const float*)d_in[4];
  const float* dec2_b   = (const float*)d_in[5];
  const float* dmean_w  = (const float*)d_in[6];
  const float* dmean_b  = (const float*)d_in[7];
  const float* dstd_w   = (const float*)d_in[8];
  const float* dstd_b   = (const float*)d_in[9];
  const float* enc0_w   = (const float*)d_in[10];
  const float* enc0_b   = (const float*)d_in[11];
  const float* enc2_w   = (const float*)d_in[12];
  const float* enc2_b   = (const float*)d_in[13];
  const float* qmean_w  = (const float*)d_in[14];
  const float* qmean_b  = (const float*)d_in[15];
  const float* qstd_w   = (const float*)d_in[16];
  const float* qstd_b   = (const float*)d_in[17];
  const float* gru_b_hh = (const float*)d_in[18];
  const float* gru_b_ih = (const float*)d_in[19];
  const float* gru_w_hh = (const float*)d_in[20];
  const float* gru_w_ih = (const float*)d_in[21];
  const float* phix_w   = (const float*)d_in[22];
  const float* phix_b   = (const float*)d_in[23];
  const float* phiz_w   = (const float*)d_in[24];
  const float* phiz_b   = (const float*)d_in[25];
  const float* pri0_w   = (const float*)d_in[26];
  const float* pri0_b   = (const float*)d_in[27];
  const float* pmean_w  = (const float*)d_in[28];
  const float* pmean_b  = (const float*)d_in[29];
  const float* pstd_w   = (const float*)d_in[30];
  const float* pstd_b   = (const float*)d_in[31];

  char* ws = (char*)d_ws;
  float*     acc = (float*)(ws + OFF_ACC);
  _Float16*  w16 = (_Float16*)(ws + OFF_W16);
  _Float16*  hs  = (_Float16*)(ws + OFF_HS);
  _Float16*  zhs = (_Float16*)(ws + OFF_ZHS);

  k_init<<<1, 64, 0, stream>>>(acc);

  struct CvtJob { const float* src; size_t off; int n; } jobs[] = {
    { dec0_w,  WO_DEC0,      131072 }, { dec2_w,  WO_DEC2,      65536 },
    { dmean_w, WO_DECMEAN,   8192   }, { dstd_w,  WO_DECSTD,    8192  },
    { enc0_w,  WO_ENC0,      196608 }, { enc2_w,  WO_ENC2,      65536 },
    { qmean_w, WO_ENCMEAN,   16384  }, { qstd_w,  WO_ENCSTD,    16384 },
    { gru_w_hh, WO_GRUWHH,   196608 }, { gru_w_ih, WO_GRUWIH,   196608 },
    { phix_w,  WO_PHIX,      8192   }, { phiz_w,  WO_PHIZ,      16384 },
    { pri0_w,  WO_PRIOR0,    131072 }, { pmean_w, WO_PRIORMEAN, 16384 },
    { pstd_w,  WO_PRIORSTD,  16384  },
  };
  for (auto& j : jobs)
    k_cvt<<<(j.n + 255) / 256, 256, 0, stream>>>(j.src, w16 + j.off, j.n);

  phase1_scan<<<M1_ / MT_, 256, 0, stream>>>(
      X, eps, w16,
      phix_b, phiz_b, pri0_b, pmean_b, pstd_b,
      enc0_b, enc2_b, qmean_b, qstd_b,
      gru_b_ih, gru_b_hh,
      hs, zhs, acc);

  phase2_dec<<<(B_ * T_ * IW_) / MT2_, 256, 0, stream>>>(
      X, w16, dec0_b, dec2_b, dmean_b, dstd_b, hs, zhs, acc);

  k_fin<<<1, 1, 0, stream>>>(acc, (float*)d_out);
}
